// TransformGroup_55645596287642
// MI455X (gfx1250) — matrix operand order repaired and checked by reading, not compiled, not run on hardware
//
#include <hip/hip_runtime.h>

// Sizes fixed by the reference
#define D_MODEL 2048
#define RANK    32
#define NUM_T   64
#define BATCH   2048
#define LDW     2112   // workspace row stride: 2048 Vx cols + 64 pre_act cols

#define TILE_M 128
#define TILE_N 64
#define TILE_K 32

typedef __attribute__((ext_vector_type(16))) __bf16 v16bf;
typedef __attribute__((ext_vector_type(8)))  float  v8f;

union Frag { v16bf v; uint4 q[2]; };

__device__ inline unsigned short bf16_rne(float f) {
  unsigned u = __float_as_uint(f);
  return (unsigned short)((u + 0x7FFFu + ((u >> 16) & 1u)) >> 16);
}
__device__ inline float bf16_up(unsigned short h) {
  return __uint_as_float(((unsigned)h) << 16);
}
__device__ inline void split2(float f, unsigned short& hi, unsigned short& lo) {
  unsigned short h = bf16_rne(f);
  hi = h;
  lo = bf16_rne(f - bf16_up(h));
}

struct STiles {
  unsigned short aHi[TILE_M][TILE_K];
  unsigned short aLo[TILE_M][TILE_K];
  unsigned short bHi[TILE_N][TILE_K];
  unsigned short bLo[TILE_N][TILE_K];
};

// C[M x N] = A[M x K] @ B[K x N], B given transposed (N-major rows).
// bMode 0: BT row c at BT + c*ldbt (covers V-cat and encoder).
// bMode 1: BT is U (64, 2048, 32); BT[d][k'=n*32+r] = U[n][d][r]; each
//          aligned 32-wide K slice is one contiguous 64x32 f32 chunk.
// fp32 emulated as bf16x3: a*b ~= aH*bH + aH*bL + aL*bH on v_wmma_f32_16x16x32_bf16.
__global__ __launch_bounds__(256)
void gemm_bf16x3(const float* __restrict__ A, int lda,
                 const float* __restrict__ BT, int ldbt,
                 float* __restrict__ C, int ldc,
                 int K, int bMode)
{
  __shared__ __align__(16) STiles s;
  const int tid   = threadIdx.x;
  const int mBase = blockIdx.y * TILE_M;
  const int nBase = blockIdx.x * TILE_N;

  const int lane = tid & 31;
  const int wave = tid >> 5;
  const int wm   = wave >> 1;   // 0..3 : 32-row strip
  const int wn   = wave & 1;    // 0..1 : 32-col strip
  const int half = lane >> 4;   // lane group hi/lo per ISA layouts
  const int ml   = lane & 15;

  v8f acc[2][2] = {};

#pragma unroll 1
  for (int kb = 0; kb < K; kb += TILE_K) {
    // ---- stage A tile (128x32 f32 -> bf16 hi/lo), 4 float4 per thread
    {
      const float* Ab = A + (size_t)mBase * lda + kb;
#pragma unroll
      for (int t = 0; t < 4; ++t) {
        int idx = tid + t * 256;          // 0..1023
        int row = idx >> 3;
        int c4  = (idx & 7) << 2;
        float4 v = *(const float4*)(Ab + (size_t)row * lda + c4);
        split2(v.x, s.aHi[row][c4+0], s.aLo[row][c4+0]);
        split2(v.y, s.aHi[row][c4+1], s.aLo[row][c4+1]);
        split2(v.z, s.aHi[row][c4+2], s.aLo[row][c4+2]);
        split2(v.w, s.aHi[row][c4+3], s.aLo[row][c4+3]);
      }
    }
    // ---- stage B tile (64x32, N-major), 2 float4 per thread
    {
#pragma unroll
      for (int t = 0; t < 2; ++t) {
        int idx = tid + t * 256;          // 0..511
        int row = idx >> 3;
        int c4  = (idx & 7) << 2;
        float4 v;
        if (bMode == 0) {
          v = *(const float4*)(BT + (size_t)(nBase + row) * ldbt + kb + c4);
        } else {
          const float* Bb = BT + (size_t)(kb >> 5) * (D_MODEL * RANK)
                               + (size_t)nBase * RANK;
          v = *(const float4*)(Bb + (size_t)idx * 4);
        }
        split2(v.x, s.bHi[row][c4+0], s.bLo[row][c4+0]);
        split2(v.y, s.bHi[row][c4+1], s.bLo[row][c4+1]);
        split2(v.z, s.bHi[row][c4+2], s.bLo[row][c4+2]);
        split2(v.w, s.bHi[row][c4+3], s.bLo[row][c4+3]);
      }
    }
    __syncthreads();

    // ---- fragments per ISA VGPR layouts (wave32, 16x16x32 bf16)
    Frag aH[2], aL[2], bH[2], bL[2];
#pragma unroll
    for (int i = 0; i < 2; ++i) {
      int r = wm * 32 + i * 16 + ml;      // A: lane holds row r; K chunks at
      aH[i].q[0] = *(const uint4*)&s.aHi[r][half * 8];        // K = half*8..+8
      aH[i].q[1] = *(const uint4*)&s.aHi[r][16 + half * 8];   // K = 16+half*8..+8
      aL[i].q[0] = *(const uint4*)&s.aLo[r][half * 8];
      aL[i].q[1] = *(const uint4*)&s.aLo[r][16 + half * 8];
    }
#pragma unroll
    for (int j = 0; j < 2; ++j) {
      int c = wn * 32 + j * 16 + ml;      // B: lane holds col c, K = half*8..+8, 16+half*8..+8
      bH[j].q[0] = *(const uint4*)&s.bHi[c][half * 8];
      bH[j].q[1] = *(const uint4*)&s.bHi[c][16 + half * 8];
      bL[j].q[0] = *(const uint4*)&s.bLo[c][half * 8];
      bL[j].q[1] = *(const uint4*)&s.bLo[c][16 + half * 8];
    }

#pragma unroll
    for (int i = 0; i < 2; ++i)
#pragma unroll
      for (int j = 0; j < 2; ++j) {
        acc[i][j] = __builtin_amdgcn_wmma_f32_16x16x32_bf16(
            false, aH[i].v, false, bH[j].v, (short)0, acc[i][j], false, false);
        acc[i][j] = __builtin_amdgcn_wmma_f32_16x16x32_bf16(
            false, aH[i].v, false, bL[j].v, (short)0, acc[i][j], false, false);
        acc[i][j] = __builtin_amdgcn_wmma_f32_16x16x32_bf16(
            false, aL[i].v, false, bH[j].v, (short)0, acc[i][j], false, false);
      }
    __syncthreads();
  }

  // ---- writeback: f32 C/D layout — VGPR v holds M = v + 8*half, N = lane&15
#pragma unroll
  for (int i = 0; i < 2; ++i)
#pragma unroll
    for (int j = 0; j < 2; ++j) {
      int col = nBase + wn * 32 + j * 16 + ml;
#pragma unroll
      for (int v = 0; v < 8; ++v) {
        int row = mBase + wm * 32 + i * 16 + half * 8 + v;
        C[(size_t)row * ldc + col] = acc[i][j][v];
      }
    }
}

// gate = relu(pre_acts - bias); write gate output; scale Vx in place -> G
__global__ __launch_bounds__(256)
void gate_kernel(float* __restrict__ ws, const float* __restrict__ bias,
                 float* __restrict__ gate_out)
{
  const int b = blockIdx.x;
  __shared__ float g[NUM_T];
  const int tid = threadIdx.x;
  if (tid < NUM_T) {
    float p  = ws[(size_t)b * LDW + D_MODEL + tid] - bias[tid];
    float gg = p > 0.0f ? p : 0.0f;   // jumprelu fwd, threshold 0
    g[tid] = gg;
    gate_out[(size_t)b * NUM_T + tid] = gg;
  }
  __syncthreads();
  for (int k = tid; k < D_MODEL; k += 256)
    ws[(size_t)b * LDW + k] *= g[k >> 5];
}

// frobenius_norms[n] = ||U_n||_F * ||V_n||_F / sqrt(D*R) = sqrt(su*sv)/256
__global__ __launch_bounds__(256)
void fro_kernel(const float* __restrict__ U, const float* __restrict__ V,
                float* __restrict__ out)
{
  const int n = blockIdx.x;
  __shared__ float red[256];
  const int tid = threadIdx.x;
  const float* Un = U + (size_t)n * (D_MODEL * RANK);
  const float* Vn = V + (size_t)n * (D_MODEL * RANK);
  float su = 0.f, sv = 0.f;
  for (int i = tid; i < D_MODEL * RANK; i += 256) {
    float a = Un[i]; su += a * a;
    float c = Vn[i]; sv += c * c;
  }
  red[tid] = su; __syncthreads();
  for (int s = 128; s > 0; s >>= 1) {
    if (tid < s) red[tid] += red[tid + s];
    __syncthreads();
  }
  float tsu = red[0]; __syncthreads();
  red[tid] = sv; __syncthreads();
  for (int s = 128; s > 0; s >>= 1) {
    if (tid < s) red[tid] += red[tid + s];
    __syncthreads();
  }
  if (tid == 0) out[n] = sqrtf(tsu * red[0]) / 256.0f;
}

extern "C" void kernel_launch(void* const* d_in, const int* in_sizes, int n_in,
                              void* d_out, int out_size, void* d_ws, size_t ws_size,
                              hipStream_t stream) {
  (void)in_sizes; (void)n_in; (void)out_size; (void)ws_size;
  const float* x    = (const float*)d_in[0];   // (2048, 2048)
  const float* V    = (const float*)d_in[1];   // (64, 32, 2048)
  const float* U    = (const float*)d_in[2];   // (64, 2048, 32)
  const float* enc  = (const float*)d_in[3];   // (64, 2048)
  const float* bias = (const float*)d_in[4];   // (64,)

  float* out  = (float*)d_out;                         // (2048, 2048)
  float* gate = out + (size_t)BATCH * D_MODEL;         // (2048, 64)
  float* fro  = gate + (size_t)BATCH * NUM_T;          // (64,)
  float* ws   = (float*)d_ws;                          // (2048 x 2112) f32, ~17.3 MB

  dim3 blk(256);
  // Vx_all = x @ Vcat^T  -> ws cols [0, 2048)
  gemm_bf16x3<<<dim3(D_MODEL / TILE_N, BATCH / TILE_M), blk, 0, stream>>>(
      x, D_MODEL, V, D_MODEL, ws, LDW, D_MODEL, 0);
  // pre_acts = x @ enc^T -> ws cols [2048, 2112)
  gemm_bf16x3<<<dim3(NUM_T / TILE_N, BATCH / TILE_M), blk, 0, stream>>>(
      x, D_MODEL, enc, D_MODEL, ws + D_MODEL, LDW, D_MODEL, 0);
  // gate + in-place scale Vx -> G
  gate_kernel<<<dim3(BATCH), blk, 0, stream>>>(ws, bias, gate);
  // output = G @ Ucat
  gemm_bf16x3<<<dim3(D_MODEL / TILE_N, BATCH / TILE_M), blk, 0, stream>>>(
      ws, LDW, U, 0, out, D_MODEL, D_MODEL, 1);
  // norms
  fro_kernel<<<dim3(NUM_T), blk, 0, stream>>>(U, V, fro);
}
